// SlimMambaBlock_79182017069613
// MI455X (gfx1250) — matrix operand order repaired and checked by reading, not compiled, not run on hardware
//
#include <hip/hip_runtime.h>
#include <hip/hip_bf16.h>

typedef __bf16 bf16;
typedef __attribute__((ext_vector_type(16))) __bf16 v16bf;
typedef __attribute__((ext_vector_type(8)))  __bf16 v8bf;
typedef __attribute__((ext_vector_type(4)))  __bf16 v4bf;
typedef __attribute__((ext_vector_type(8)))  float  v8f;

#define B_     16
#define K_     2048
#define D_     512
#define INNER_ 1024
#define M_     (B_ * K_)   // 32768 rows
#define KW_    5
#define EPS_   1e-5f

static __device__ __forceinline__ v8f wmma_bf16(v16bf a, v16bf b, v8f c) {
  // D = A(16x32 bf16) x B(32x16 bf16) + C(16x16 f32)
  return __builtin_amdgcn_wmma_f32_16x16x32_bf16(false, a, false, b, (short)0, c,
                                                 false, false);
}

// ---------------------------------------------------------------------------
// Kernel 0: f32 -> bf16 weight conversion
// ---------------------------------------------------------------------------
__global__ void k_f32_to_bf16(const float* __restrict__ s, bf16* __restrict__ d, int n) {
  int i = blockIdx.x * blockDim.x + threadIdx.x;
  if (i < n) d[i] = (bf16)s[i];
}

// ---------------------------------------------------------------------------
// Kernel 1: RMSNorm rows of x[M,D] -> h bf16.  128 threads/row, 4 elems/thread.
// ---------------------------------------------------------------------------
__global__ __launch_bounds__(128) void k_rmsnorm(const float* __restrict__ x,
                                                 const float* __restrict__ w,
                                                 bf16* __restrict__ h) {
  const int row = blockIdx.x;
  const int t = threadIdx.x;
  const float4 v = reinterpret_cast<const float4*>(x + (size_t)row * D_)[t];
  float ss = v.x * v.x + v.y * v.y + v.z * v.z + v.w * v.w;
#pragma unroll
  for (int off = 16; off > 0; off >>= 1) ss += __shfl_xor(ss, off);
  __shared__ float part[4];
  if ((t & 31) == 0) part[t >> 5] = ss;
  __syncthreads();
  const float tot = part[0] + part[1] + part[2] + part[3];
  const float scale = rsqrtf(tot * (1.0f / (float)D_) + EPS_);
  const float4 wv = reinterpret_cast<const float4*>(w)[t];
  v4bf o;
  o[0] = (bf16)(v.x * scale * wv.x);
  o[1] = (bf16)(v.y * scale * wv.y);
  o[2] = (bf16)(v.z * scale * wv.z);
  o[3] = (bf16)(v.w * scale * wv.w);
  reinterpret_cast<v4bf*>(h + (size_t)row * D_)[t] = o;
}

// ---------------------------------------------------------------------------
// Kernel 2: dual GEMM  a=h*Wa^T, b=h*Wb^T  (M x 512) x (512 x 1024)
// 4 waves/block, each wave: one 16-row M-tile x 64 cols, both outputs.
// ---------------------------------------------------------------------------
__global__ __launch_bounds__(128) void k_gemm_ab(const bf16* __restrict__ h,
                                                 const bf16* __restrict__ Wa,
                                                 const bf16* __restrict__ Wb,
                                                 bf16* __restrict__ aout,
                                                 bf16* __restrict__ bout) {
  const int lane = threadIdx.x & 31;
  const int wave = threadIdx.x >> 5;
  const int r = lane & 15, half = lane >> 4;
  const int mtile = blockIdx.x * 4 + wave;   // 0..2047
  const int row0 = mtile * 16;
  const int n0 = blockIdx.y * 64;            // 0..960

  v8f accA[4] = {};
  v8f accB[4] = {};
  const bf16* ap = h + (size_t)(row0 + r) * D_ + half * 16;

  for (int k0 = 0; k0 < D_; k0 += 32) {
    v16bf Af;
    Af = *reinterpret_cast<const v16bf*>(ap + k0);
#pragma unroll
    for (int nt = 0; nt < 4; ++nt) {
      const size_t nrow = (size_t)(n0 + nt * 16 + r);
      v16bf Ba = *reinterpret_cast<const v16bf*>(Wa + nrow * D_ + k0 + half * 16);
      accA[nt] = wmma_bf16(Af, Ba, accA[nt]);
      v16bf Bb = *reinterpret_cast<const v16bf*>(Wb + nrow * D_ + k0 + half * 16);
      accB[nt] = wmma_bf16(Af, Bb, accB[nt]);
    }
  }
#pragma unroll
  for (int nt = 0; nt < 4; ++nt) {
#pragma unroll
    for (int j = 0; j < 8; ++j) {
      const size_t rr = (size_t)(row0 + j + half * 8);
      const size_t cc = (size_t)(n0 + nt * 16 + r);
      aout[rr * INNER_ + cc] = (bf16)accA[nt][j];
      bout[rr * INNER_ + cc] = (bf16)accB[nt][j];
    }
  }
}

// ---------------------------------------------------------------------------
// Kernel 3: fused depthwise conv(5) + bias + SiLU + linear scan + gate.
// One thread per (batch, channel); rolling 5-tap window; writes g over a_pre
// (reads always >= 3 rows ahead of writes -> alias safe).
// ---------------------------------------------------------------------------
__global__ __launch_bounds__(256) void k_conv_scan(const bf16* apre,
                                                   const bf16* __restrict__ bmat,
                                                   const float* __restrict__ convw,
                                                   const float* __restrict__ convb,
                                                   const float* __restrict__ alpha,
                                                   const float* __restrict__ beta,
                                                   const float* __restrict__ gamma,
                                                   const float* __restrict__ delta,
                                                   bf16* gout) {
  const int b = blockIdx.x >> 2;
  const int i = (blockIdx.x & 3) * 256 + threadIdx.x;   // channel 0..1023
  const float cw0 = convw[i * KW_ + 0], cw1 = convw[i * KW_ + 1];
  const float cw2 = convw[i * KW_ + 2], cw3 = convw[i * KW_ + 3];
  const float cw4 = convw[i * KW_ + 4];
  const float cb = convb[i];
  const float asig = 1.0f / (1.0f + __expf(-alpha[i]));
  const float bt = beta[i], gm = gamma[i], dl = delta[i];

  const bf16* ap = apre + (size_t)b * K_ * INNER_ + i;
  const bf16* bp = bmat + (size_t)b * K_ * INNER_ + i;
  bf16* gp = gout + (size_t)b * K_ * INNER_ + i;

  float w0 = 0.f, w1 = 0.f;
  float w2 = (float)ap[0];
  float w3 = (float)ap[(size_t)1 * INNER_];
  float w4 = (float)ap[(size_t)2 * INNER_];
  float s = 0.f;

  for (int k = 0; k < K_; ++k) {
    float u = cw0 * w0 + cw1 * w1 + cw2 * w2 + cw3 * w3 + cw4 * w4 + cb;
    u = u / (1.0f + __expf(-u));          // SiLU
    s = asig * s + bt * u;                // scan
    const float y = gm * s + dl * u;
    const float bv = (float)bp[(size_t)k * INNER_];
    gp[(size_t)k * INNER_] = (bf16)(y * bv);
    w0 = w1; w1 = w2; w2 = w3; w3 = w4;
    w4 = (k + 3 < K_) ? (float)ap[(size_t)(k + 3) * INNER_] : 0.f;
  }
}

// ---------------------------------------------------------------------------
// Kernel 4: output GEMM  y = g[M,1024] * Wout^T[1024,512]  + residual x.
// ---------------------------------------------------------------------------
__global__ __launch_bounds__(128) void k_gemm_out(const bf16* __restrict__ g,
                                                  const bf16* __restrict__ Wo,
                                                  const float* __restrict__ x,
                                                  float* __restrict__ out) {
  const int lane = threadIdx.x & 31;
  const int wave = threadIdx.x >> 5;
  const int r = lane & 15, half = lane >> 4;
  const int mtile = blockIdx.x * 4 + wave;
  const int row0 = mtile * 16;
  const int n0 = blockIdx.y * 64;            // 0..448

  v8f acc[4] = {};
  const bf16* gp = g + (size_t)(row0 + r) * INNER_ + half * 16;

  for (int k0 = 0; k0 < INNER_; k0 += 32) {
    v16bf Af;
    Af = *reinterpret_cast<const v16bf*>(gp + k0);
#pragma unroll
    for (int nt = 0; nt < 4; ++nt) {
      const size_t nrow = (size_t)(n0 + nt * 16 + r);
      v16bf Bf = *reinterpret_cast<const v16bf*>(Wo + nrow * INNER_ + k0 + half * 16);
      acc[nt] = wmma_bf16(Af, Bf, acc[nt]);
    }
  }
#pragma unroll
  for (int nt = 0; nt < 4; ++nt) {
#pragma unroll
    for (int j = 0; j < 8; ++j) {
      const size_t rr = (size_t)(row0 + j + half * 8);
      const size_t cc = (size_t)(n0 + nt * 16 + r);
      out[rr * D_ + cc] = x[rr * D_ + cc] + acc[nt][j];
    }
  }
}

// ---------------------------------------------------------------------------
extern "C" void kernel_launch(void* const* d_in, const int* in_sizes, int n_in,
                              void* d_out, int out_size, void* d_ws, size_t ws_size,
                              hipStream_t stream) {
  const float* x     = (const float*)d_in[0];
  const float* normw = (const float*)d_in[1];
  const float* Wa    = (const float*)d_in[2];
  const float* Wb    = (const float*)d_in[3];
  const float* convw = (const float*)d_in[4];
  const float* convb = (const float*)d_in[5];
  const float* alpha = (const float*)d_in[6];
  const float* beta  = (const float*)d_in[7];
  const float* gamma = (const float*)d_in[8];
  const float* delta = (const float*)d_in[9];
  const float* Wout  = (const float*)d_in[10];
  float* out = (float*)d_out;

  char* ws = (char*)d_ws;
  size_t off = 0;
  bf16* hb   = (bf16*)(ws + off); off += (size_t)M_ * D_ * sizeof(bf16);       // 32 MB
  bf16* abuf = (bf16*)(ws + off); off += (size_t)M_ * INNER_ * sizeof(bf16);   // 64 MB (a_pre, then g)
  bf16* bbuf = (bf16*)(ws + off); off += (size_t)M_ * INNER_ * sizeof(bf16);   // 64 MB
  bf16* WaB  = (bf16*)(ws + off); off += (size_t)INNER_ * D_ * sizeof(bf16);
  bf16* WbB  = (bf16*)(ws + off); off += (size_t)INNER_ * D_ * sizeof(bf16);
  bf16* WoB  = (bf16*)(ws + off); off += (size_t)D_ * INNER_ * sizeof(bf16);
  if (off > ws_size) return;

  const int nw = INNER_ * D_;
  k_f32_to_bf16<<<(nw + 255) / 256, 256, 0, stream>>>(Wa, WaB, nw);
  k_f32_to_bf16<<<(nw + 255) / 256, 256, 0, stream>>>(Wb, WbB, nw);
  k_f32_to_bf16<<<(nw + 255) / 256, 256, 0, stream>>>(Wout, WoB, nw);

  k_rmsnorm<<<M_, 128, 0, stream>>>(x, normw, hb);

  k_gemm_ab<<<dim3(M_ / 64, INNER_ / 64), 128, 0, stream>>>(hb, WaB, WbB, abuf, bbuf);

  k_conv_scan<<<B_ * (INNER_ / 256), 256, 0, stream>>>(abuf, bbuf, convw, convb,
                                                       alpha, beta, gamma, delta, abuf);

  k_gemm_out<<<dim3(M_ / 64, D_ / 64), 128, 0, stream>>>(abuf, WoB, x, out);
}
